// Block_34746285425248
// MI455X (gfx1250) — matrix operand order repaired and checked by reading, not compiled, not run on hardware
//
#include <hip/hip_runtime.h>
#include <hip/hip_bf16.h>
#include <math.h>

// ---------------------------------------------------------------------------
// Transformer block (pre-LN, causal MHA + GELU MLP) for MI455X / gfx1250.
// All GEMMs + attention matmuls via v_wmma_f32_16x16x32_bf16 (wave32 WMMA),
// f32 accumulate, f32 LayerNorm/residual path.
// ---------------------------------------------------------------------------

typedef __bf16 bf16_t;
typedef __bf16 v8bf  __attribute__((ext_vector_type(8)));
typedef __bf16 v16bf __attribute__((ext_vector_type(16)));
typedef float  v8f   __attribute__((ext_vector_type(8)));

#define D_   1024
#define H_   16
#define DK_  64
#define T_   2048
#define B_   4
#define M_   (B_ * T_)     // 8192 tokens
#define FF_  (4 * D_)      // 4096
#define EPS_ 1e-5f

// ---------------------------------------------------------------------------
// WMMA helpers.
// A fragment (16x32 bf16, MxK): lane(<16)=row M, half=lane>>4.
//   elems 0..7  <-> K = kb + half*8 + (0..7)        (VGPR 0..3)
//   elems 8..15 <-> K = kb + 16 + half*8 + (0..7)   (VGPR 4..7)
// B fragment (32x16 bf16, KxN) from column-major storage Bt[N][K]:
//   N = n0 + lane%16, elems 0..7 <-> K = kb + half*8 + (0..7),
//   elems 8..15 <-> K = kb + 16 + half*8 + (0..7)
// C/D (16x16 f32): elem j <-> row M = j + half*8, col N = lane%16.
// ---------------------------------------------------------------------------

__device__ __forceinline__ v8f wmma_bf16(v16bf a, v16bf b, v8f c) {
  return __builtin_amdgcn_wmma_f32_16x16x32_bf16(false, a, false, b,
                                                 (short)0, c, false, false);
}

__device__ __forceinline__ v16bf load_frag_a(const bf16_t* base, int ld,
                                             int row0, int kb, int l16, int half) {
  const bf16_t* p = base + (size_t)(row0 + l16) * ld + kb + half * 8;
  v8bf lo = *reinterpret_cast<const v8bf*>(p);
  v8bf hi = *reinterpret_cast<const v8bf*>(p + 16);
  v16bf r;
#pragma unroll
  for (int i = 0; i < 8; ++i) { r[i] = lo[i]; r[i + 8] = hi[i]; }
  return r;
}

__device__ __forceinline__ v16bf load_frag_b(const bf16_t* baseT, int ld,
                                             int n0, int kb, int l16, int half) {
  const bf16_t* p = baseT + (size_t)(n0 + l16) * ld + kb + half * 8;
  v8bf lo = *reinterpret_cast<const v8bf*>(p);
  v8bf hi = *reinterpret_cast<const v8bf*>(p + 16);
  v16bf r;
#pragma unroll
  for (int i = 0; i < 8; ++i) { r[i] = lo[i]; r[i + 8] = hi[i]; }
  return r;
}

__device__ __forceinline__ v8f vzero8() {
  v8f z = {0.f, 0.f, 0.f, 0.f, 0.f, 0.f, 0.f, 0.f};
  return z;
}

// ---------------------------------------------------------------------------
// Tiled transpose + f32 -> bf16 convert:  in [batch][R][C] -> out [batch][C][R]
// ---------------------------------------------------------------------------
__global__ __launch_bounds__(256)
void transpose_cvt_kernel(const float* __restrict__ in, bf16_t* __restrict__ out,
                          int R, int C) {
  __shared__ float tile[32][33];
  const int batch = blockIdx.z;
  const float* ip = in + (size_t)batch * R * C;
  bf16_t* op = out + (size_t)batch * R * C;
  const int c0 = blockIdx.x * 32, r0 = blockIdx.y * 32;
  const int tx = threadIdx.x, ty = threadIdx.y;  // (32, 8)
#pragma unroll
  for (int k = 0; k < 32; k += 8)
    tile[ty + k][tx] = ip[(size_t)(r0 + ty + k) * C + (c0 + tx)];
  __syncthreads();
#pragma unroll
  for (int k = 0; k < 32; k += 8)
    op[(size_t)(c0 + ty + k) * R + (r0 + tx)] = (bf16_t)tile[tx][ty + k];
}

// ---------------------------------------------------------------------------
// LayerNorm (f32 in) -> bf16 out. One 256-thread block per token row.
// ---------------------------------------------------------------------------
__global__ __launch_bounds__(256)
void layernorm_kernel(const float* __restrict__ x, const float* __restrict__ g,
                      const float* __restrict__ be, bf16_t* __restrict__ out) {
  __shared__ float s1[256], s2[256];
  const int row = blockIdx.x;
  const float* xr = x + (size_t)row * D_;
  float a = 0.f, b = 0.f;
  for (int i = threadIdx.x; i < D_; i += 256) {
    float v = xr[i];
    a += v; b += v * v;
  }
  s1[threadIdx.x] = a; s2[threadIdx.x] = b;
  __syncthreads();
  for (int off = 128; off > 0; off >>= 1) {
    if (threadIdx.x < off) {
      s1[threadIdx.x] += s1[threadIdx.x + off];
      s2[threadIdx.x] += s2[threadIdx.x + off];
    }
    __syncthreads();
  }
  const float mu = s1[0] * (1.0f / D_);
  const float var = s2[0] * (1.0f / D_) - mu * mu;
  const float rs = rsqrtf(var + EPS_);
  for (int i = threadIdx.x; i < D_; i += 256) {
    float v = (xr[i] - mu) * rs * g[i] + be[i];
    out[(size_t)row * D_ + i] = (bf16_t)v;
  }
}

// ---------------------------------------------------------------------------
// Generic bf16 WMMA GEMM:  C[M,N] = A[M,K] (row-major) x B (col-major as Bt[N,K])
// 256 threads = 8 waves (4 M-waves x 2 N-waves); 32x64 tile per wave
// (2 A-frags x 4 B-frags -> 8 WMMAs per 6 fragment loads per K-step);
// block tile 128x128.
// Epilogue modes:
//   0: store bf16 row-major [M,N]
//   1: store bf16 V-transposed [B,H,DK,T]      (for attention V)
//   2: f32 out = acc + bias[n] + resid[m,n]    (residual adds)
//   3: bf16 out = gelu(acc + bias[n])          (FFN hidden)
// ---------------------------------------------------------------------------
template <int MODE>
__global__ __launch_bounds__(256)
void gemm_wmma_kernel(const bf16_t* __restrict__ A, const bf16_t* __restrict__ Bt,
                      int M, int N, int K,
                      const float* __restrict__ bias,
                      const float* __restrict__ resid,
                      void* __restrict__ out) {
  const int tid  = threadIdx.x;
  const int wave = tid >> 5;
  const int lane = tid & 31;
  const int l16  = lane & 15;
  const int half = lane >> 4;
  const int wm = wave & 3, wn = wave >> 2;
  const int m0 = blockIdx.x * 128 + wm * 32;
  const int n0 = blockIdx.y * 128 + wn * 64;

  v8f acc[2][4];
#pragma unroll
  for (int mi = 0; mi < 2; ++mi)
#pragma unroll
    for (int ni = 0; ni < 4; ++ni) acc[mi][ni] = vzero8();

  for (int kb = 0; kb < K; kb += 32) {
    v16bf a0 = load_frag_a(A, K, m0, kb, l16, half);
    v16bf a1 = load_frag_a(A, K, m0 + 16, kb, l16, half);
    v16bf b0 = load_frag_b(Bt, K, n0, kb, l16, half);
    v16bf b1 = load_frag_b(Bt, K, n0 + 16, kb, l16, half);
    v16bf b2 = load_frag_b(Bt, K, n0 + 32, kb, l16, half);
    v16bf b3 = load_frag_b(Bt, K, n0 + 48, kb, l16, half);
    acc[0][0] = wmma_bf16(a0, b0, acc[0][0]);
    acc[0][1] = wmma_bf16(a0, b1, acc[0][1]);
    acc[0][2] = wmma_bf16(a0, b2, acc[0][2]);
    acc[0][3] = wmma_bf16(a0, b3, acc[0][3]);
    acc[1][0] = wmma_bf16(a1, b0, acc[1][0]);
    acc[1][1] = wmma_bf16(a1, b1, acc[1][1]);
    acc[1][2] = wmma_bf16(a1, b2, acc[1][2]);
    acc[1][3] = wmma_bf16(a1, b3, acc[1][3]);
  }

#pragma unroll
  for (int mi = 0; mi < 2; ++mi) {
#pragma unroll
    for (int ni = 0; ni < 4; ++ni) {
#pragma unroll
      for (int j = 0; j < 8; ++j) {
        const int m = m0 + mi * 16 + j + half * 8;
        const int n = n0 + ni * 16 + l16;
        const float v = acc[mi][ni][j];
        if constexpr (MODE == 0) {
          ((bf16_t*)out)[(size_t)m * N + n] = (bf16_t)v;
        } else if constexpr (MODE == 1) {
          const int b = m >> 11, t = m & (T_ - 1);
          const int hh = n >> 6, kk = n & (DK_ - 1);
          ((bf16_t*)out)[((size_t)((b * H_ + hh) * DK_ + kk)) * T_ + t] = (bf16_t)v;
        } else if constexpr (MODE == 2) {
          ((float*)out)[(size_t)m * N + n] = v + bias[n] + resid[(size_t)m * N + n];
        } else {
          float u = v + bias[n];
          u = 0.5f * u * (1.0f + erff(u * 0.70710678118654752f));  // exact GELU
          ((bf16_t*)out)[(size_t)m * N + n] = (bf16_t)u;
        }
      }
    }
  }
}

// ---------------------------------------------------------------------------
// Causal FlashAttention. One wave per 16-query tile; key chunks of 32.
// Q,K stored bf16 [B,T,H*DK]; V stored transposed bf16 [B,H,DK,T].
// Output O bf16 [B,T,H*DK].
// ---------------------------------------------------------------------------
__global__ __launch_bounds__(128)
void attention_kernel(const bf16_t* __restrict__ Q, const bf16_t* __restrict__ Kb,
                      const bf16_t* __restrict__ Vt, bf16_t* __restrict__ O) {
  __shared__ __align__(16) bf16_t Pbuf[4][16 * 32];  // per-wave P tile
  const int wave = threadIdx.x >> 5;
  const int lane = threadIdx.x & 31;
  const int l16  = lane & 15;
  const int half = lane >> 4;
  const int row_off = half * 8;

  const int tileid = blockIdx.x * 4 + wave;   // 8192 tiles total
  const int qt = tileid & 127;                // T/16 = 128 query tiles
  const int bh = tileid >> 7;                 // 0..63
  const int b = bh >> 4, h = bh & 15;

  const bf16_t* qbase = Q + ((size_t)(b * T_ + qt * 16)) * D_ + h * DK_;
  const bf16_t* kbase = Kb + ((size_t)(b * T_)) * D_ + h * DK_;
  const bf16_t* vbase = Vt + ((size_t)((b * H_ + h) * DK_)) * T_;
  bf16_t* pb = Pbuf[wave];

  // q fragments for the whole tile (DK = 64 = 2 K-chunks)
  const v16bf aq0 = load_frag_a(qbase, D_, 0, 0, l16, half);
  const v16bf aq1 = load_frag_a(qbase, D_, 0, 32, l16, half);

  v8f o0 = vzero8(), o1 = vzero8(), o2 = vzero8(), o3 = vzero8();
  float mrow[8], lrow[8];
#pragma unroll
  for (int j = 0; j < 8; ++j) { mrow[j] = -1.0e30f; lrow[j] = 0.f; }

  const int qlast = qt * 16 + 15;
  for (int s0 = 0; s0 <= qlast; s0 += 32) {
    // ---- S = q @ K^T for 32 keys (two 16-col accumulators) ----
    v8f sa0 = vzero8(), sa1 = vzero8();
    {
      v16bf bk;
      bk = load_frag_b(kbase, D_, s0, 0, l16, half);       sa0 = wmma_bf16(aq0, bk, sa0);
      bk = load_frag_b(kbase, D_, s0, 32, l16, half);      sa0 = wmma_bf16(aq1, bk, sa0);
      bk = load_frag_b(kbase, D_, s0 + 16, 0, l16, half);  sa1 = wmma_bf16(aq0, bk, sa1);
      bk = load_frag_b(kbase, D_, s0 + 16, 32, l16, half); sa1 = wmma_bf16(aq1, bk, sa1);
    }
    // ---- scale + causal mask ----
#pragma unroll
    for (int j = 0; j < 8; ++j) {
      const int qg = qt * 16 + j + row_off;
      float v0 = sa0[j] * 0.125f;           // DK^-0.5
      float v1 = sa1[j] * 0.125f;
      if (s0 + l16 > qg)      v0 = -3.0e38f;
      if (s0 + 16 + l16 > qg) v1 = -3.0e38f;
      sa0[j] = v0; sa1[j] = v1;
    }
    // ---- online softmax: per-row max / exp / sum (width-16 reductions) ----
#pragma unroll
    for (int j = 0; j < 8; ++j) {
      float cm = fmaxf(sa0[j], sa1[j]);
#pragma unroll
      for (int off = 1; off < 16; off <<= 1) cm = fmaxf(cm, __shfl_xor(cm, off, 16));
      const float mnew = fmaxf(mrow[j], cm);
      const float scl = __expf(mrow[j] - mnew);
      const float e0 = __expf(sa0[j] - mnew);
      const float e1 = __expf(sa1[j] - mnew);
      float rs = e0 + e1;
#pragma unroll
      for (int off = 1; off < 16; off <<= 1) rs += __shfl_xor(rs, off, 16);
      mrow[j] = mnew;
      lrow[j] = lrow[j] * scl + rs;
      o0[j] *= scl; o1[j] *= scl; o2[j] *= scl; o3[j] *= scl;
      const int prow = j + row_off;
      pb[prow * 32 + l16]      = (bf16_t)e0;   // C-layout -> LDS row-major
      pb[prow * 32 + 16 + l16] = (bf16_t)e1;
    }
    // Wave-private LDS: per-wave DS ops are in-order; keep compiler from
    // hoisting the loads above the stores and drain the DS pipe.
    asm volatile("s_wait_dscnt 0" ::: "memory");
    // ---- reload P as an A fragment (16x32) ----
    v16bf ap;
    {
      const bf16_t* pp = pb + l16 * 32 + half * 8;
      v8bf lo = *reinterpret_cast<const v8bf*>(pp);
      v8bf hi = *reinterpret_cast<const v8bf*>(pp + 16);
#pragma unroll
      for (int i = 0; i < 8; ++i) { ap[i] = lo[i]; ap[i + 8] = hi[i]; }
    }
    // ---- O += P @ V  (V pre-transposed: rows are head-dim, cols are keys) ----
    v16bf bv;
    bv = load_frag_b(vbase, T_, 0,  s0, l16, half); o0 = wmma_bf16(ap, bv, o0);
    bv = load_frag_b(vbase, T_, 16, s0, l16, half); o1 = wmma_bf16(ap, bv, o1);
    bv = load_frag_b(vbase, T_, 32, s0, l16, half); o2 = wmma_bf16(ap, bv, o2);
    bv = load_frag_b(vbase, T_, 48, s0, l16, half); o3 = wmma_bf16(ap, bv, o3);
  }

  // ---- normalize and write O (concat-head layout [B,T,H*DK]) ----
  bf16_t* obase = O + ((size_t)(b * T_ + qt * 16)) * D_ + h * DK_;
#pragma unroll
  for (int j = 0; j < 8; ++j) {
    const float inv = 1.0f / lrow[j];
    const int row = j + row_off;
    obase[(size_t)row * D_ + 0  + l16] = (bf16_t)(o0[j] * inv);
    obase[(size_t)row * D_ + 16 + l16] = (bf16_t)(o1[j] * inv);
    obase[(size_t)row * D_ + 32 + l16] = (bf16_t)(o2[j] * inv);
    obase[(size_t)row * D_ + 48 + l16] = (bf16_t)(o3[j] * inv);
  }
}

// ---------------------------------------------------------------------------
// Host-side orchestration.
// ---------------------------------------------------------------------------
extern "C" void kernel_launch(void* const* d_in, const int* in_sizes, int n_in,
                              void* d_out, int out_size, void* d_ws, size_t ws_size,
                              hipStream_t stream) {
  const float* x   = (const float*)d_in[0];
  const float* Wq  = (const float*)d_in[1];
  const float* Wk  = (const float*)d_in[2];
  const float* Wv  = (const float*)d_in[3];
  const float* Wo  = (const float*)d_in[4];
  const float* bo  = (const float*)d_in[5];
  const float* W1  = (const float*)d_in[6];
  const float* b1  = (const float*)d_in[7];
  const float* W2  = (const float*)d_in[8];
  const float* b2  = (const float*)d_in[9];
  const float* g1  = (const float*)d_in[10];
  const float* be1 = (const float*)d_in[11];
  const float* g2  = (const float*)d_in[12];
  const float* be2 = (const float*)d_in[13];
  (void)in_sizes; (void)n_in; (void)out_size; (void)ws_size;

  char* ws = (char*)d_ws;
  const size_t MB = (size_t)1 << 20;
  bf16_t* h   = (bf16_t*)(ws + 0);         // 16 MiB  (reused for h2)
  bf16_t* wqt = (bf16_t*)(ws + 16 * MB);   //  2 MiB  [H*DK, D]
  bf16_t* wkt = (bf16_t*)(ws + 18 * MB);   //  2 MiB
  bf16_t* wvt = (bf16_t*)(ws + 20 * MB);   //  2 MiB
  bf16_t* wot = (bf16_t*)(ws + 22 * MB);   //  2 MiB  [D, D] (transposed)
  bf16_t* w1t = (bf16_t*)(ws + 24 * MB);   //  8 MiB  [4D, D]
  bf16_t* w2t = (bf16_t*)(ws + 32 * MB);   //  8 MiB  [D, 4D]
  bf16_t* qb  = (bf16_t*)(ws + 40 * MB);   // 16 MiB  [B,T,H*DK]
  bf16_t* kb  = (bf16_t*)(ws + 56 * MB);   // 16 MiB  [B,T,H*DK]
  bf16_t* vt  = (bf16_t*)(ws + 72 * MB);   // 16 MiB  [B,H,DK,T]
  bf16_t* ob  = (bf16_t*)(ws + 88 * MB);   // 16 MiB  [B,T,H*DK]
  float*  x2  = (float*)(ws + 104 * MB);   // 32 MiB  [B,T,D] f32
  bf16_t* ffh = qb;  // 64 MiB alias over q/k/v/o (dead after Wo projection)

  const dim3 tb(32, 8);
  // Weight prep: transpose+convert to bf16 column-major.
  transpose_cvt_kernel<<<dim3(DK_ / 32, D_ / 32, H_), tb, 0, stream>>>(Wq, wqt, D_, DK_);
  transpose_cvt_kernel<<<dim3(DK_ / 32, D_ / 32, H_), tb, 0, stream>>>(Wk, wkt, D_, DK_);
  transpose_cvt_kernel<<<dim3(DK_ / 32, D_ / 32, H_), tb, 0, stream>>>(Wv, wvt, D_, DK_);
  transpose_cvt_kernel<<<dim3(D_ / 32, D_ / 32, 1), tb, 0, stream>>>(Wo, wot, D_, D_);
  transpose_cvt_kernel<<<dim3(FF_ / 32, D_ / 32, 1), tb, 0, stream>>>(W1, w1t, D_, FF_);
  transpose_cvt_kernel<<<dim3(D_ / 32, FF_ / 32, 1), tb, 0, stream>>>(W2, w2t, FF_, D_);

  // LN1 -> h (bf16)
  layernorm_kernel<<<M_, 256, 0, stream>>>(x, g1, be1, h);

  // QKV projections (each: 8192x1024 @ 1024x1024), block tile 128x128
  gemm_wmma_kernel<0><<<dim3(M_ / 128, D_ / 128), 256, 0, stream>>>(
      h, wqt, M_, D_, D_, nullptr, nullptr, qb);
  gemm_wmma_kernel<0><<<dim3(M_ / 128, D_ / 128), 256, 0, stream>>>(
      h, wkt, M_, D_, D_, nullptr, nullptr, kb);
  gemm_wmma_kernel<1><<<dim3(M_ / 128, D_ / 128), 256, 0, stream>>>(
      h, wvt, M_, D_, D_, nullptr, nullptr, vt);

  // Causal flash attention: 8192 query tiles, 4 waves per block.
  attention_kernel<<<(B_ * H_ * (T_ / 16)) / 4, 128, 0, stream>>>(qb, kb, vt, ob);

  // Output projection + residual: x2 = x + O @ Wo + bo  (f32)
  gemm_wmma_kernel<2><<<dim3(M_ / 128, D_ / 128), 256, 0, stream>>>(
      ob, wot, M_, D_, D_, bo, x, x2);

  // LN2 -> h2 (reuse h buffer)
  layernorm_kernel<<<M_, 256, 0, stream>>>(x2, g2, be2, h);

  // FFN: hidden = gelu(h2 @ W1 + b1) (bf16), out = x2 + hidden @ W2 + b2 (f32)
  gemm_wmma_kernel<3><<<dim3(M_ / 128, FF_ / 128), 256, 0, stream>>>(
      h, w1t, M_, FF_, D_, b1, nullptr, ffh);
  gemm_wmma_kernel<2><<<dim3(M_ / 128, D_ / 128), 256, 0, stream>>>(
      ffh, w2t, M_, D_, FF_, b2, x2, (float*)d_out);
}
